// ColumnSelfAttention_21002390077476
// MI455X (gfx1250) — matrix operand order repaired and checked by reading, not compiled, not run on hardware
//
#include <hip/hip_runtime.h>
#include <hip/hip_bf16.h>

// ---------------------------------------------------------------------------
// Column self-attention for MI455X (gfx1250), wave32 + WMMA f16->f32.
//   EMBED=768, HEADS=12, DK=64, R=128, C=256, B=1
// Pipeline:
//   K0: one-time f32->f16 convert of x and the 4 weight matrices (ws).
//   K1: q/k/v = x16 @ W16^T + b. M=256/N=64 block tile, 32x64 per wave.
//       Tiles staged global->LDS with GLOBAL_LOAD_ASYNC_TO_LDS_B128
//       (double-buffered, s_wait_asynccnt partial waits) -> zero VGPR
//       staging, no spills. q scaled dk^-0.5; q,k stored [h][c][row][d],
//       v stored transposed [h][c][d][row] (f16).
//   K2: per (h,c): S = q k^T (WMMA), masked softmax (fp32, shfl reductions),
//       probs -> d_out (fp32), P staged f16 in LDS, ctx = P v (WMMA).
//   K3: out = ctx @ Wo16^T + bo (same async-staged GEMM), fp32 -> d_out.
// ---------------------------------------------------------------------------

#define EMBED 768
#define HEADS 12
#define DKH   64
#define RDIM  128
#define CDIM  256
#define TOKENS (RDIM * CDIM)

typedef __attribute__((ext_vector_type(16))) _Float16 v16h;
typedef __attribute__((ext_vector_type(8)))  _Float16 h8;
typedef __attribute__((ext_vector_type(4)))  _Float16 h4;
typedef __attribute__((ext_vector_type(8)))  float    v8f;

__device__ __forceinline__ v8f wmma_f16(v16h a, v16h b, v8f c) {
    return __builtin_amdgcn_wmma_f32_16x16x32_f16(
        /*neg_a=*/false, a, /*neg_b=*/false, b,
        /*c_mod=*/(short)0, c, /*reuse_a=*/false, /*reuse_b=*/false);
}

// Low 32 bits of a flat pointer to LDS == LDS byte offset (ISA: LDS aperture
// maps addr[31:0] to LDS address).
__device__ __forceinline__ unsigned lds_off(const void* p) {
    return (unsigned)(size_t)p;
}

// CDNA5 async global->LDS copy, 16B per lane, tracked by ASYNCcnt.
__device__ __forceinline__ void async_b128(unsigned lds, const _Float16* g) {
    asm volatile("global_load_async_to_lds_b128 %0, %1, off"
                 :: "v"(lds), "v"(g) : "memory");
}
__device__ __forceinline__ void wait_async_le5() {
    asm volatile("s_wait_asynccnt 0x5" ::: "memory");
}
__device__ __forceinline__ void wait_async_0() {
    asm volatile("s_wait_asynccnt 0x0" ::: "memory");
}

// A fragment (16x32, f16), row-major source with leading dim `ld` (halves).
// ISA layout (CDNA5 ISA): lane m=l&15; lanes 0-15 hold K in [0,8)+[16,24),
// lanes 16-31 hold K in [8,16)+[24,32). Two 16B contiguous chunks per lane.
__device__ __forceinline__ v16h load_frag_a(const _Float16* __restrict__ src,
                                            int ld, int lane) {
    const int m  = lane & 15;
    const int kh = (lane >> 4) << 3;            // 0 or 8
    const _Float16* p = src + (size_t)m * ld + kh;
    h8 lo = *(const h8*)(p);                    // K = kh .. kh+7
    h8 hi = *(const h8*)(p + 16);               // K = kh+16 .. kh+23
    v16h a;
#pragma unroll
    for (int j = 0; j < 8; ++j) { a[j] = lo[j]; a[8 + j] = hi[j]; }
    return a;
}

// B fragment (32x16, f16) from source laid out [n][k] row-major, ld in halves.
// Lane n=l&15; lanes 0-15 hold K [0,8)+[16,24), lanes 16-31 hold K [8,16)+[24,32).
__device__ __forceinline__ v16h load_frag_b(const _Float16* __restrict__ src,
                                            int ld, int lane) {
    const int n  = lane & 15;
    const int kb = (lane >> 4) << 3;            // 0 or 8
    const _Float16* p = src + (size_t)n * ld + kb;
    h8 lo = *(const h8*)(p);
    h8 hi = *(const h8*)(p + 16);
    v16h b;
#pragma unroll
    for (int j = 0; j < 8; ++j) { b[j] = lo[j]; b[8 + j] = hi[j]; }
    return b;
}

// ------------------------------ Kernel 0 -----------------------------------
// grid-stride f32 -> f16 convert (float4 -> h4), n4 = element_count/4
__global__ __launch_bounds__(256)
void cvt_f16_kernel(const float* __restrict__ src, _Float16* __restrict__ dst,
                    int n4) {
    int idx    = blockIdx.x * 256 + threadIdx.x;
    int stride = gridDim.x * 256;
    for (; idx < n4; idx += stride) {
        float4 f = *(const float4*)(src + (size_t)idx * 4);
        h4 hv; hv[0] = (_Float16)f.x; hv[1] = (_Float16)f.y;
               hv[2] = (_Float16)f.z; hv[3] = (_Float16)f.w;
        *(h4*)(dst + (size_t)idx * 4) = hv;
    }
}

// ------------------------------ Kernel 1 -----------------------------------
// grid.x = TOKENS/256 token tiles, grid.y = 36 (3 proj * 12 n-tiles of 64).
// Block tile M=256 x N=64; each wave computes 32 rows x 64 cols (8 WMMA/step).
__global__ __launch_bounds__(256)
void qkv_proj_kernel(const _Float16* __restrict__ x16,
                     const _Float16* __restrict__ Wq16,
                     const _Float16* __restrict__ Wk16,
                     const _Float16* __restrict__ Wv16,
                     const float* __restrict__ bq, const float* __restrict__ bk,
                     const float* __restrict__ bv,
                     _Float16* __restrict__ qout, _Float16* __restrict__ kout,
                     _Float16* __restrict__ vtout) {
    __shared__ _Float16 As[2][256][40];   // double-buffered, 2 x 20480 B
    __shared__ _Float16 Bs[2][64][40];    // double-buffered, 2 x  5120 B

    const int tid  = threadIdx.x;
    const int lane = tid & 31;
    const int wave = tid >> 5;
    const int t0   = blockIdx.x * 256;
    const int gy   = blockIdx.y;
    const int proj = gy / 12;                 // 0=q 1=k 2=v
    const int n0   = (gy % 12) * 64;

    const _Float16* W = (proj == 0) ? Wq16 : (proj == 1) ? Wk16 : Wv16;
    const float* bias = (proj == 0) ? bq   : (proj == 1) ? bk   : bv;

    // per-thread tile-copy coordinates (A: 1024 16B chunks -> 4/thread, B: 1)
    const int arow = tid >> 2, aseg = (tid & 3) * 8;
    const int brow = tid >> 2, bseg = (tid & 3) * 8;

    const _Float16* xa = x16 + (size_t)(t0 + arow) * EMBED + aseg;
    const _Float16* wb = W   + (size_t)(n0 + brow) * EMBED + bseg;

    unsigned ldsA[4], ldsB;
#pragma unroll
    for (int it = 0; it < 4; ++it) ldsA[it] = lds_off(&As[0][arow + it * 64][aseg]);
    ldsB = lds_off(&Bs[0][brow][bseg]);
    const unsigned strA = (unsigned)sizeof(_Float16) * 256 * 40;  // 20480
    const unsigned strB = (unsigned)sizeof(_Float16) * 64 * 40;   //  5120

    v8f acc[2][4] = {};

    // prologue: async-stage tile 0 into buffer 0
#pragma unroll
    for (int it = 0; it < 4; ++it) async_b128(ldsA[it], xa + (size_t)it * 64 * EMBED);
    async_b128(ldsB, wb);

    int cur = 0;
    for (int k0 = 0; k0 < EMBED; k0 += 32, cur ^= 1) {
        if (k0 + 32 < EMBED) {     // async-stage next tile into other buffer
            const unsigned nb = cur ^ 1;
#pragma unroll
            for (int it = 0; it < 4; ++it)
                async_b128(ldsA[it] + nb * strA,
                           xa + (size_t)it * 64 * EMBED + k0 + 32);
            async_b128(ldsB + nb * strB, wb + k0 + 32);
            wait_async_le5();      // current tile's 5 copies complete
        } else {
            wait_async_0();
        }
        __syncthreads();

        // load ALL fragments first, then back-to-back WMMAs (partial dscnt waits)
        v16h af0 = load_frag_a(&As[cur][wave * 32][0],      40, lane);
        v16h af1 = load_frag_a(&As[cur][wave * 32 + 16][0], 40, lane);
        v16h bf[4];
#pragma unroll
        for (int nt = 0; nt < 4; ++nt)
            bf[nt] = load_frag_b(&Bs[cur][nt * 16][0], 40, lane);
#pragma unroll
        for (int nt = 0; nt < 4; ++nt) {
            acc[0][nt] = wmma_f16(af0, bf[nt], acc[0][nt]);
            acc[1][nt] = wmma_f16(af1, bf[nt], acc[1][nt]);
        }
        __syncthreads();           // buffer free for the next async stage
    }

    // epilogue: bias (+scale for q), scatter to attention-friendly layouts
    const float scaling = 0.125f;              // DK^-0.5
    const int nlane = lane & 15;
#pragma unroll
    for (int mb = 0; mb < 2; ++mb) {
#pragma unroll
        for (int nt = 0; nt < 4; ++nt) {
            int n = n0 + nt * 16 + nlane;
            float bb = bias[n];
            int h = n >> 6, d = n & 63;
#pragma unroll
            for (int r = 0; r < 8; ++r) {
                int m = mb * 16 + ((lane >> 4) << 3) + r;   // C-layout row
                int t = t0 + wave * 32 + m;
                int i = t / CDIM, c = t % CDIM;
                float val = acc[mb][nt][r] + bb;
                size_t hc = (size_t)h * CDIM + c;
                if (proj == 0) {
                    qout[(hc * RDIM + i) * DKH + d] = (_Float16)(val * scaling);
                } else if (proj == 1) {
                    kout[(hc * RDIM + i) * DKH + d] = (_Float16)val;
                } else {
                    vtout[(hc * DKH + d) * RDIM + i] = (_Float16)val;  // transposed
                }
            }
        }
    }
}

// ------------------------------ Kernel 2 -----------------------------------
// grid.x = HEADS*CDIM; one workgroup computes 128x128 attention for (h,c).
__global__ __launch_bounds__(256)
void attn_kernel(const _Float16* __restrict__ q, const _Float16* __restrict__ k,
                 const _Float16* __restrict__ vt,
                 const unsigned char* __restrict__ pmask,
                 float* __restrict__ probs_out, _Float16* __restrict__ ctx_out) {
    __shared__ _Float16 Ps[128][136];  // probs f16, padded rows (272B, 16B-aligned)
    __shared__ float    msk[128];

    const int tid  = threadIdx.x;
    const int lane = tid & 31;
    const int wave = tid >> 5;
    const int hc   = blockIdx.x;
    const int h    = hc / CDIM;
    const int c    = hc % CDIM;

    const _Float16* qb = q  + (size_t)hc * RDIM * DKH;
    const _Float16* kb = k  + (size_t)hc * RDIM * DKH;
    const _Float16* vb = vt + (size_t)hc * DKH * RDIM;

    if (tid < 128) msk[tid] = pmask[(size_t)tid * CDIM + c] ? 1.0f : 0.0f;
    __syncthreads();

    const int i0 = wave * 16;
    // Q fragments for this wave's 16 rows (two K halves of DK=64)
    v16h qf0 = load_frag_a(qb + (size_t)i0 * DKH,      DKH, lane);
    v16h qf1 = load_frag_a(qb + (size_t)i0 * DKH + 32, DKH, lane);

    v8f sc[8];
#pragma unroll
    for (int jt = 0; jt < 8; ++jt) {
        v16h kf0 = load_frag_b(kb + (size_t)(jt * 16) * DKH,      DKH, lane);
        v16h kf1 = load_frag_b(kb + (size_t)(jt * 16) * DKH + 32, DKH, lane);
        v8f a = {};
        a = wmma_f16(qf0, kf0, a);
        a = wmma_f16(qf1, kf1, a);
        sc[jt] = a;
    }

    // masked softmax over j (row-wise). C-layout: lane holds col j=jt*16+(l&15),
    // rows m=8*(l>>4)+r -> reduce across the 16-lane half via shfl_xor.
    const int nlane = lane & 15;
    float rowmax[8], rowsum[8];
#pragma unroll
    for (int r = 0; r < 8; ++r) {
        float mx = -3.0e38f;
#pragma unroll
        for (int jt = 0; jt < 8; ++jt) {
            int j = jt * 16 + nlane;
            float s = sc[jt][r];
            if (msk[j] != 0.0f) s = -10000.0f;   // where(mask, -1e4, s)
            sc[jt][r] = s;
            mx = fmaxf(mx, s);
        }
#pragma unroll
        for (int off = 1; off < 16; off <<= 1)
            mx = fmaxf(mx, __shfl_xor(mx, off, 32));
        rowmax[r] = mx;
    }
#pragma unroll
    for (int r = 0; r < 8; ++r) {
        float sum = 0.0f;
#pragma unroll
        for (int jt = 0; jt < 8; ++jt) {
            float e = __expf(sc[jt][r] - rowmax[r]);
            sc[jt][r] = e;
            sum += e;
        }
#pragma unroll
        for (int off = 1; off < 16; off <<= 1)
            sum += __shfl_xor(sum, off, 32);
        rowsum[r] = sum;
    }

    // write probs (fp32 output) and stage P (f16) for ctx GEMM
    float* pb = probs_out + (size_t)hc * RDIM * RDIM;
#pragma unroll
    for (int jt = 0; jt < 8; ++jt) {
#pragma unroll
        for (int r = 0; r < 8; ++r) {
            int m = ((lane >> 4) << 3) + r;
            int j = jt * 16 + nlane;
            float p = sc[jt][r] / rowsum[r];
            pb[(size_t)(i0 + m) * RDIM + j] = p;
            Ps[i0 + m][j] = (_Float16)p;
        }
    }
    __syncthreads();

    // ctx = P @ v : rows i0..i0+15, 64 cols, K=128 in 4 steps of 32
    v8f acc[4] = {};
#pragma unroll
    for (int kk = 0; kk < 4; ++kk) {
        v16h af = load_frag_a(&Ps[i0][kk * 32], 136, lane);
        v16h bf[4];
#pragma unroll
        for (int nt = 0; nt < 4; ++nt)
            bf[nt] = load_frag_b(vb + (size_t)(nt * 16) * RDIM + kk * 32, RDIM, lane);
#pragma unroll
        for (int nt = 0; nt < 4; ++nt)
            acc[nt] = wmma_f16(af, bf[nt], acc[nt]);
    }

    // ctx stored [token][e] f16 for the output GEMM (A-matrix row-major)
#pragma unroll
    for (int nt = 0; nt < 4; ++nt) {
        int d = nt * 16 + nlane;
#pragma unroll
        for (int r = 0; r < 8; ++r) {
            int m = ((lane >> 4) << 3) + r;
            size_t t = (size_t)(i0 + m) * CDIM + c;
            ctx_out[t * EMBED + h * DKH + d] = (_Float16)acc[nt][r];
        }
    }
}

// ------------------------------ Kernel 3 -----------------------------------
// grid.x = TOKENS/256, grid.y = EMBED/64.  M=256 x N=64 per block, 32x64/wave.
__global__ __launch_bounds__(256)
void out_proj_kernel(const _Float16* __restrict__ ctx,
                     const _Float16* __restrict__ Wo16,
                     const float* __restrict__ bo,
                     float* __restrict__ out) {
    __shared__ _Float16 As[2][256][40];
    __shared__ _Float16 Bs[2][64][40];

    const int tid  = threadIdx.x;
    const int lane = tid & 31;
    const int wave = tid >> 5;
    const int t0   = blockIdx.x * 256;
    const int n0   = blockIdx.y * 64;

    const int arow = tid >> 2, aseg = (tid & 3) * 8;
    const int brow = tid >> 2, bseg = (tid & 3) * 8;

    const _Float16* ca = ctx  + (size_t)(t0 + arow) * EMBED + aseg;
    const _Float16* wb = Wo16 + (size_t)(n0 + brow) * EMBED + bseg;

    unsigned ldsA[4], ldsB;
#pragma unroll
    for (int it = 0; it < 4; ++it) ldsA[it] = lds_off(&As[0][arow + it * 64][aseg]);
    ldsB = lds_off(&Bs[0][brow][bseg]);
    const unsigned strA = (unsigned)sizeof(_Float16) * 256 * 40;
    const unsigned strB = (unsigned)sizeof(_Float16) * 64 * 40;

    v8f acc[2][4] = {};

#pragma unroll
    for (int it = 0; it < 4; ++it) async_b128(ldsA[it], ca + (size_t)it * 64 * EMBED);
    async_b128(ldsB, wb);

    int cur = 0;
    for (int k0 = 0; k0 < EMBED; k0 += 32, cur ^= 1) {
        if (k0 + 32 < EMBED) {
            const unsigned nb = cur ^ 1;
#pragma unroll
            for (int it = 0; it < 4; ++it)
                async_b128(ldsA[it] + nb * strA,
                           ca + (size_t)it * 64 * EMBED + k0 + 32);
            async_b128(ldsB + nb * strB, wb + k0 + 32);
            wait_async_le5();
        } else {
            wait_async_0();
        }
        __syncthreads();

        v16h af0 = load_frag_a(&As[cur][wave * 32][0],      40, lane);
        v16h af1 = load_frag_a(&As[cur][wave * 32 + 16][0], 40, lane);
        v16h bf[4];
#pragma unroll
        for (int nt = 0; nt < 4; ++nt)
            bf[nt] = load_frag_b(&Bs[cur][nt * 16][0], 40, lane);
#pragma unroll
        for (int nt = 0; nt < 4; ++nt) {
            acc[0][nt] = wmma_f16(af0, bf[nt], acc[0][nt]);
            acc[1][nt] = wmma_f16(af1, bf[nt], acc[1][nt]);
        }
        __syncthreads();
    }

    const int nlane = lane & 15;
#pragma unroll
    for (int mb = 0; mb < 2; ++mb) {
#pragma unroll
        for (int nt = 0; nt < 4; ++nt) {
            int n = n0 + nt * 16 + nlane;
            float bb = bo[n];
#pragma unroll
            for (int r = 0; r < 8; ++r) {
                int m = mb * 16 + ((lane >> 4) << 3) + r;
                int t = t0 + wave * 32 + m;
                out[(size_t)t * EMBED + n] = acc[mb][nt][r] + bb;
            }
        }
    }
}

// ---------------------------------------------------------------------------
extern "C" void kernel_launch(void* const* d_in, const int* in_sizes, int n_in,
                              void* d_out, int out_size, void* d_ws, size_t ws_size,
                              hipStream_t stream) {
    const float*         x  = (const float*)d_in[0];
    const unsigned char* pm = (const unsigned char*)d_in[1];  // bool mask (all zero)
    const float* Wq = (const float*)d_in[2]; const float* bq = (const float*)d_in[3];
    const float* Wk = (const float*)d_in[4]; const float* bk = (const float*)d_in[5];
    const float* Wv = (const float*)d_in[6]; const float* bv = (const float*)d_in[7];
    const float* Wo = (const float*)d_in[8]; const float* bo = (const float*)d_in[9];

    float* out   = (float*)d_out;                       // [T, 768] fp32
    float* probs = out + (size_t)TOKENS * EMBED;        // [H, C, 1, R, R] fp32

    const size_t BUF  = (size_t)TOKENS * EMBED;         // f16 elements
    const size_t WSZ  = (size_t)EMBED * EMBED;
    _Float16* qbuf  = (_Float16*)d_ws;                  // [h][c][i][d]
    _Float16* kbuf  = qbuf  + BUF;                      // [h][c][j][d]
    _Float16* vtbuf = kbuf  + BUF;                      // [h][c][d][j]
    _Float16* cbuf  = vtbuf + BUF;                      // [t][e]
    _Float16* x16   = cbuf  + BUF;                      // [t][e] f16 copy of x
    _Float16* Wq16  = x16   + BUF;
    _Float16* Wk16  = Wq16  + WSZ;
    _Float16* Wv16  = Wk16  + WSZ;
    _Float16* Wo16  = Wv16  + WSZ;
    (void)in_sizes; (void)n_in; (void)out_size; (void)ws_size;

    dim3 blk(256);
    // K0: one-time f32 -> f16 conversions
    cvt_f16_kernel<<<dim3(2048), blk, 0, stream>>>(x,  x16,  (int)(BUF / 4));
    cvt_f16_kernel<<<dim3(288),  blk, 0, stream>>>(Wq, Wq16, (int)(WSZ / 4));
    cvt_f16_kernel<<<dim3(288),  blk, 0, stream>>>(Wk, Wk16, (int)(WSZ / 4));
    cvt_f16_kernel<<<dim3(288),  blk, 0, stream>>>(Wv, Wv16, (int)(WSZ / 4));
    cvt_f16_kernel<<<dim3(288),  blk, 0, stream>>>(Wo, Wo16, (int)(WSZ / 4));

    qkv_proj_kernel<<<dim3(TOKENS / 256, 36), blk, 0, stream>>>(
        x16, Wq16, Wk16, Wv16, bq, bk, bv, qbuf, kbuf, vtbuf);
    attn_kernel<<<dim3(HEADS * CDIM), blk, 0, stream>>>(
        qbuf, kbuf, vtbuf, pm, probs, cbuf);
    out_proj_kernel<<<dim3(TOKENS / 256, EMBED / 64), blk, 0, stream>>>(
        cbuf, Wo16, bo, out);
}
